// MultiheadAttention_84524956385852
// MI455X (gfx1250) — matrix operand order repaired and checked by reading, not compiled, not run on hardware
//
#include <hip/hip_runtime.h>

typedef __attribute__((ext_vector_type(16))) __bf16 v16bf;
typedef __attribute__((ext_vector_type(8)))  __bf16 bf16x8;
typedef __attribute__((ext_vector_type(4)))  __bf16 bf16x4;
typedef __attribute__((ext_vector_type(8)))  float  v8f;
typedef __attribute__((ext_vector_type(4)))  float  f32x4;
typedef __attribute__((ext_vector_type(4)))  int    v4i;

#define AS1 __attribute__((address_space(1)))
#define AS3 __attribute__((address_space(3)))

static constexpr int Bc = 2, Sc = 2048, Dc = 1024, Hc = 16, DHc = 64, NFc = 4;
static constexpr int Mc = Bc * Sc;  // 4096

#if __has_builtin(__builtin_amdgcn_global_load_async_to_lds_b128)
#define HAVE_ASYNC_LDS 1
#else
#define HAVE_ASYNC_LDS 0
#endif

__device__ __forceinline__ void wait_asynccnt0() {
#if __has_builtin(__builtin_amdgcn_s_wait_asynccnt)
  __builtin_amdgcn_s_wait_asynccnt(0);
#else
  asm volatile("s_wait_asynccnt 0x0" ::: "memory");
#endif
}

#if HAVE_ASYNC_LDS
__device__ __forceinline__ void async_copy_b128(const __bf16* gsrc, __bf16* ldst) {
  __builtin_amdgcn_global_load_async_to_lds_b128(
      (AS1 v4i*)(v4i*)(gsrc), (AS3 v4i*)(v4i*)(ldst), 0, 0);
}
#endif

// ---------------------------------------------------------------------------
// fp32 -> bf16 conversion, 4 elements per thread
// ---------------------------------------------------------------------------
__global__ __launch_bounds__(256) void convert_f32_bf16(
    const float* __restrict__ in, __bf16* __restrict__ out, int n) {
  int i = (blockIdx.x * blockDim.x + threadIdx.x) * 4;
  if (i < n) {
    f32x4 v = *(const f32x4*)(in + i);
    bf16x4 o;
    o[0] = (__bf16)v[0]; o[1] = (__bf16)v[1];
    o[2] = (__bf16)v[2]; o[3] = (__bf16)v[3];
    *(bf16x4*)(out + i) = o;
  }
}

// ---------------------------------------------------------------------------
// Weff[o,d] = sum_f w[f] * Wout[f,o,d] (bf16), beff[d] = sum_f w[f]*bout[f,d]
// ---------------------------------------------------------------------------
__global__ __launch_bounds__(256) void weff_build(
    const float* __restrict__ w, const float* __restrict__ Wout,
    const float* __restrict__ bout, __bf16* __restrict__ Weff,
    float* __restrict__ beff) {
  const int i = blockIdx.x * blockDim.x + threadIdx.x;
  const int DD = Dc * Dc;
  if (i < DD) {
    float s = 0.f;
#pragma unroll
    for (int f = 0; f < NFc; ++f) s += w[f] * Wout[(size_t)f * DD + i];
    Weff[i] = (__bf16)s;
  }
  if (i < Dc) {
    float s = 0.f;
#pragma unroll
    for (int f = 0; f < NFc; ++f) s += w[f] * bout[f * Dc + i];
    beff[i] = s;
  }
}

// ---------------------------------------------------------------------------
// Generic GEMM: C[m,n] = sum_k A[m,k] * Bw[n,k] + bias[n]
//   A: Mtot x K bf16 row-major, Bw: N x K bf16 row-major (torch Linear weight)
//   MODE 0: bf16 row-major out; MODE 1: bf16 per-head transposed [b,h,dh,s];
//   MODE 2: fp32 row-major out.
// Block = 256 threads (8 waves). Wave computes 16(M) x 64(N). B tile
// (64n x 32k) is double-buffered in LDS via GLOBAL_LOAD_ASYNC_TO_LDS_B128:
// the k+32 tile streams into the alternate buffer while WMMAs consume tile k.
// ---------------------------------------------------------------------------
template <int MODE>
__global__ __launch_bounds__(256) void gemm_wmma_bf16(
    const __bf16* __restrict__ A, const __bf16* __restrict__ Bw,
    const float* __restrict__ bias, void* __restrict__ Cout,
    int Mtot, int N, int K) {
  __shared__ __align__(16) __bf16 lds_b[2][64 * 32];  // [buf][n_local][k] 8 KB

  const int tid  = threadIdx.x;
  const int lane = tid & 31;
  const int wave = tid >> 5;
  const int half = (lane < 16) ? 0 : 1;
  const int ln   = lane & 15;

  const int mtile = blockIdx.y * 128 + wave * 16;
  const int nbase = blockIdx.x * 64;

  v8f acc[4];
#pragma unroll
  for (int f = 0; f < 4; ++f)
#pragma unroll
    for (int r = 0; r < 8; ++r) acc[f][r] = 0.f;

  const __bf16* arow = A + (size_t)(mtile + ln) * K;
  const int a_k0 = half * 8;

  // per-thread source/dest for the cooperative 64x32 B tile (16B each)
  const __bf16* bsrc = Bw + (size_t)(nbase + (tid >> 2)) * K + (tid & 3) * 8;
  __bf16* bdst0 = &lds_b[0][(tid >> 2) * 32 + (tid & 3) * 8];
  __bf16* bdst1 = &lds_b[1][(tid >> 2) * 32 + (tid & 3) * 8];

#if HAVE_ASYNC_LDS
  async_copy_b128(bsrc, bdst0);
#endif

  int buf = 0;
  for (int kb = 0; kb < K; kb += 32) {
#if HAVE_ASYNC_LDS
    wait_asynccnt0();      // our slice of tile `buf` has landed in LDS
    __syncthreads();       // everyone's slice has landed; prev-buf reads done
    if (kb + 32 < K) {     // prefetch next tile into the other buffer
      async_copy_b128(bsrc + kb + 32, buf ? bdst0 : bdst1);
    }
#else
    __syncthreads();
    *(bf16x8*)(buf ? bdst1 : bdst0) = *(const bf16x8*)(bsrc + kb);
    __syncthreads();
#endif

    // A fragment: lane holds M=ln, K = a_k0..a_k0+7 and a_k0+16..a_k0+23
    v16bf af;
    *(bf16x8*)&af     = *(const bf16x8*)(arow + kb + a_k0);
    ((bf16x8*)&af)[1] = *(const bf16x8*)(arow + kb + a_k0 + 16);

    const __bf16* ltile = lds_b[buf];
#pragma unroll
    for (int f = 0; f < 4; ++f) {
      const __bf16* brow = ltile + (f * 16 + ln) * 32 + half * 8;
      v16bf bf_;
      *(bf16x8*)&bf_     = *(const bf16x8*)(brow);
      ((bf16x8*)&bf_)[1] = *(const bf16x8*)(brow + 16);
      acc[f] = __builtin_amdgcn_wmma_f32_16x16x32_bf16(
          false, af, false, bf_, (short)0, acc[f], false, false);
    }
    buf ^= 1;
  }

#pragma unroll
  for (int f = 0; f < 4; ++f) {
    const int n = nbase + f * 16 + ln;
    const float bb = bias[n];
    if (MODE == 0) {
      __bf16* C = (__bf16*)Cout;
#pragma unroll
      for (int r = 0; r < 8; ++r) {
        const int m = mtile + half * 8 + r;
        C[(size_t)m * N + n] = (__bf16)(acc[f][r] + bb);
      }
    } else if (MODE == 2) {
      float* C = (float*)Cout;
#pragma unroll
      for (int r = 0; r < 8; ++r) {
        const int m = mtile + half * 8 + r;
        C[(size_t)m * N + n] = acc[f][r] + bb;
      }
    } else {  // MODE 1: per-head transposed vT[((b*H+h)*DH+dh)*S + s]
      __bf16* C = (__bf16*)Cout;
      const int h = n >> 6, dh = n & 63;
      const int bidx = mtile / Sc;
      const int s0 = mtile - bidx * Sc + half * 8;
      bf16x8 pack;
#pragma unroll
      for (int r = 0; r < 8; ++r) pack[r] = (__bf16)(acc[f][r] + bb);
      *(bf16x8*)(C + ((size_t)(bidx * Hc + h) * DHc + dh) * Sc + s0) = pack;
    }
  }
}

// ---------------------------------------------------------------------------
// Flash-style causal attention. One wave per 16-query block of one (b,h).
// q,k: [B,S,D] bf16 (head h in cols h*64..); vT: [B,H,DH,S] bf16;
// o: [B,S,D] bf16. scale = 1/sqrt(DH).
// ---------------------------------------------------------------------------
__global__ __launch_bounds__(128) void attn_wmma(
    const __bf16* __restrict__ q, const __bf16* __restrict__ k,
    const __bf16* __restrict__ vT, __bf16* __restrict__ o) {
  __shared__ __align__(16) __bf16 p_lds_all[4 * 16 * 32];  // 1 KB per wave

  const int tid  = threadIdx.x;
  const int lane = tid & 31;
  const int wave = tid >> 5;
  const int half = (lane < 16) ? 0 : 1;
  const int ln   = lane & 15;
  __bf16* p_lds = p_lds_all + wave * (16 * 32);

  const int bh = blockIdx.y;          // b*H + h
  const int b  = bh >> 4;             // H = 16
  const int h  = bh & 15;
  const int qbase = (blockIdx.x * 4 + wave) * 16;
  const float scale = 0.125f;         // 1/sqrt(64)

  // q fragments (K-dim = DH = 64 -> two K=32 fragments), loaded once
  const __bf16* qrow = q + ((size_t)(b * Sc + qbase + ln)) * Dc + h * DHc;
  v16bf qf[2];
#pragma unroll
  for (int ks = 0; ks < 2; ++ks) {
    const __bf16* p = qrow + ks * 32 + half * 8;
    *(bf16x8*)&qf[ks]     = *(const bf16x8*)(p);
    ((bf16x8*)&qf[ks])[1] = *(const bf16x8*)(p + 16);
  }

  v8f acc[4];
  float m_r[8], l_r[8];
#pragma unroll
  for (int f = 0; f < 4; ++f)
#pragma unroll
    for (int r = 0; r < 8; ++r) acc[f][r] = 0.f;
#pragma unroll
  for (int r = 0; r < 8; ++r) { m_r[r] = -1e30f; l_r[r] = 0.f; }

  for (int jbase = 0; jbase < qbase + 16; jbase += 32) {
    // ---- scores: 16 queries x 32 keys ----
    v8f c0, c1;
#pragma unroll
    for (int r = 0; r < 8; ++r) { c0[r] = 0.f; c1[r] = 0.f; }
#pragma unroll
    for (int ks = 0; ks < 2; ++ks) {
      const __bf16* kp0 =
          k + ((size_t)(b * Sc + jbase + ln)) * Dc + h * DHc + ks * 32 + half * 8;
      v16bf kf0, kf1;
      *(bf16x8*)&kf0     = *(const bf16x8*)(kp0);
      ((bf16x8*)&kf0)[1] = *(const bf16x8*)(kp0 + 16);
      const __bf16* kp1 = kp0 + (size_t)16 * Dc;
      *(bf16x8*)&kf1     = *(const bf16x8*)(kp1);
      ((bf16x8*)&kf1)[1] = *(const bf16x8*)(kp1 + 16);
      c0 = __builtin_amdgcn_wmma_f32_16x16x32_bf16(
          false, qf[ks], false, kf0, (short)0, c0, false, false);
      c1 = __builtin_amdgcn_wmma_f32_16x16x32_bf16(
          false, qf[ks], false, kf1, (short)0, c1, false, false);
    }

    // ---- online softmax over the 16x32 tile ----
    const int key0 = jbase + ln;
    float p0[8], p1[8], corr[8];
#pragma unroll
    for (int r = 0; r < 8; ++r) {
      const int qidx = qbase + half * 8 + r;
      float s0 = c0[r] * scale; if (key0 > qidx)      s0 = -1e30f;
      float s1 = c1[r] * scale; if (key0 + 16 > qidx) s1 = -1e30f;
      float vmax = fmaxf(s0, s1);
      vmax = fmaxf(vmax, __shfl_xor(vmax, 1, 16));
      vmax = fmaxf(vmax, __shfl_xor(vmax, 2, 16));
      vmax = fmaxf(vmax, __shfl_xor(vmax, 4, 16));
      vmax = fmaxf(vmax, __shfl_xor(vmax, 8, 16));
      const float mn = fmaxf(m_r[r], vmax);
      const float e0 = __expf(s0 - mn);
      const float e1 = __expf(s1 - mn);
      float rs = e0 + e1;
      rs += __shfl_xor(rs, 1, 16);
      rs += __shfl_xor(rs, 2, 16);
      rs += __shfl_xor(rs, 4, 16);
      rs += __shfl_xor(rs, 8, 16);
      const float c = __expf(m_r[r] - mn);
      l_r[r] = l_r[r] * c + rs;
      m_r[r] = mn;
      corr[r] = c;
      p0[r] = e0; p1[r] = e1;
    }
#pragma unroll
    for (int f = 0; f < 4; ++f)
#pragma unroll
      for (int r = 0; r < 8; ++r) acc[f][r] *= corr[r];

    // ---- C-layout -> A-layout via per-wave LDS tile (16 rows x 32 keys) ----
#pragma unroll
    for (int r = 0; r < 8; ++r) {
      const int row = half * 8 + r;
      p_lds[row * 32 + ln]      = (__bf16)p0[r];
      p_lds[row * 32 + 16 + ln] = (__bf16)p1[r];
    }
    v16bf pf;  // A-fragment of P (same-wave LDS ops are in order)
    {
      const __bf16* pr = p_lds + ln * 32 + half * 8;
      *(bf16x8*)&pf     = *(const bf16x8*)(pr);
      ((bf16x8*)&pf)[1] = *(const bf16x8*)(pr + 16);
    }

    // ---- PV: acc(16x64) += P(16x32keys) @ v(32keys x 64dh) ----
#pragma unroll
    for (int f = 0; f < 4; ++f) {
      const __bf16* vp =
          vT + ((size_t)(bh * DHc + f * 16 + ln)) * Sc + jbase + half * 8;
      v16bf vf;
      *(bf16x8*)&vf     = *(const bf16x8*)(vp);
      ((bf16x8*)&vf)[1] = *(const bf16x8*)(vp + 16);
      acc[f] = __builtin_amdgcn_wmma_f32_16x16x32_bf16(
          false, pf, false, vf, (short)0, acc[f], false, false);
    }
  }

  // ---- finalize: divide by l, store bf16 [B,S,D] ----
#pragma unroll
  for (int r = 0; r < 8; ++r) l_r[r] = 1.f / l_r[r];
  __bf16* orow = o + ((size_t)(b * Sc + qbase)) * Dc + h * DHc;
#pragma unroll
  for (int f = 0; f < 4; ++f)
#pragma unroll
    for (int r = 0; r < 8; ++r) {
      const int row = half * 8 + r;
      orow[(size_t)row * Dc + f * 16 + ln] = (__bf16)(acc[f][r] * l_r[r]);
    }
}

// ---------------------------------------------------------------------------
extern "C" void kernel_launch(void* const* d_in, const int* in_sizes, int n_in,
                              void* d_out, int out_size, void* d_ws,
                              size_t ws_size, hipStream_t stream) {
  const float* x       = (const float*)d_in[0];
  const float* weights = (const float*)d_in[1];
  const float* Wq      = (const float*)d_in[2];
  const float* bq      = (const float*)d_in[3];
  const float* Wk      = (const float*)d_in[4];
  const float* bk      = (const float*)d_in[5];
  const float* Wv      = (const float*)d_in[6];
  const float* bv      = (const float*)d_in[7];
  const float* Wout    = (const float*)d_in[8];
  const float* bout    = (const float*)d_in[9];

  const size_t MD = (size_t)Mc * Dc;  // 4M elems
  const size_t DD = (size_t)Dc * Dc;  // 1M elems

  __bf16* wsb   = (__bf16*)d_ws;
  __bf16* x_bf  = wsb;            // 4M
  __bf16* q_bf  = x_bf + MD;      // 4M
  __bf16* k_bf  = q_bf + MD;      // 4M
  __bf16* vT_bf = k_bf + MD;      // 4M  ([B,H,DH,S])
  __bf16* Wq_bf = vT_bf + MD;     // 1M
  __bf16* Wk_bf = Wq_bf + DD;     // 1M
  __bf16* Wv_bf = Wk_bf + DD;     // 1M
  __bf16* We_bf = Wv_bf + DD;     // 1M
  float*  beff  = (float*)(We_bf + DD);  // D floats
  __bf16* ao_bf = x_bf;           // reuse x_bf after projections

  // 1) conversions + effective output weight
  convert_f32_bf16<<<(int)(MD / 4 + 255) / 256, 256, 0, stream>>>(x, x_bf, (int)MD);
  convert_f32_bf16<<<(int)(DD / 4 + 255) / 256, 256, 0, stream>>>(Wq, Wq_bf, (int)DD);
  convert_f32_bf16<<<(int)(DD / 4 + 255) / 256, 256, 0, stream>>>(Wk, Wk_bf, (int)DD);
  convert_f32_bf16<<<(int)(DD / 4 + 255) / 256, 256, 0, stream>>>(Wv, Wv_bf, (int)DD);
  weff_build<<<(int)(DD + 255) / 256, 256, 0, stream>>>(weights, Wout, bout, We_bf, beff);

  // 2) Q/K/V projections (V stored per-head transposed)
  dim3 ggrid(Dc / 64, Mc / 128);  // (16, 32)
  gemm_wmma_bf16<0><<<ggrid, 256, 0, stream>>>(x_bf, Wq_bf, bq, q_bf, Mc, Dc, Dc);
  gemm_wmma_bf16<0><<<ggrid, 256, 0, stream>>>(x_bf, Wk_bf, bk, k_bf, Mc, Dc, Dc);
  gemm_wmma_bf16<1><<<ggrid, 256, 0, stream>>>(x_bf, Wv_bf, bv, vT_bf, Mc, Dc, Dc);

  // 3) causal flash attention
  dim3 agrid(Sc / 16 / 4, Bc * Hc);  // (32, 32)
  attn_wmma<<<agrid, 128, 0, stream>>>(q_bf, k_bf, vT_bf, ao_bf);

  // 4) output projection with combined weights -> fp32 d_out
  gemm_wmma_bf16<2><<<ggrid, 256, 0, stream>>>(ao_bf, We_bf, beff, d_out, Mc, Dc, Dc);
}
